// GraphSAGENodePredictor_12850542150153
// MI455X (gfx1250) — matrix operand order repaired and checked by reading, not compiled, not run on hardware
//
#include <hip/hip_runtime.h>
#include <hip/hip_bf16.h>

typedef __attribute__((ext_vector_type(16))) _Float16 v16h;
typedef __attribute__((ext_vector_type(8)))  float    v8f;

#define NND   100000
#define NEDG  1600000
#define INCH  128
#define HID   64
#define BNEPS 1e-5f

// ---------------------------------------------------------------- utilities
__global__ void zero_f32(float* __restrict__ p, int n) {
  int i = blockIdx.x * blockDim.x + threadIdx.x;
  int stride = gridDim.x * blockDim.x;
  for (; i < n; i += stride) p[i] = 0.0f;
}

__global__ void degree_kernel(const int* __restrict__ tgt, float* __restrict__ deg) {
  int e = blockIdx.x * blockDim.x + threadIdx.x;
  if (e < NEDG) atomicAdd(&deg[tgt[e]], 1.0f);
}

__global__ void deginv_kernel(float* __restrict__ deg) {
  int i = blockIdx.x * blockDim.x + threadIdx.x;
  if (i < NND) deg[i] = 1.0f / fmaxf(deg[i], 1.0f);
}

// ---------------------------------------------------------------- fused dual WMMA GEMM
// Yl[M x 64] = X[M x K] @ Wl,  Yr[M x 64] = X[M x K] @ Wr   (single read of X).
// One wave computes a 16-row tile x both 64-wide outputs (8 WMMA col-tiles).
// Both weight matrices are converted to f16 in LDS, pre-swizzled into per-lane
// B-fragment order: fragment (kb, f), lane L holds 16 contiguous halves
//   = K rows kb*32 + (L>=16)*8 + {0..7, 16..23}, column (f&3)*16 + (L&15),
//   weight matrix selected by f>>2 (0 = Wl, 1 = Wr).
template <int K>
__global__ __launch_bounds__(256) void gemm2_wmma(const float* __restrict__ X,
                                                  const float* __restrict__ Wl,
                                                  const float* __restrict__ Wr,
                                                  float* __restrict__ Yl,
                                                  float* __restrict__ Yr) {
  __shared__ __align__(32) _Float16 sW[2 * K * HID];
  for (int idx = threadIdx.x; idx < 2 * K * HID; idx += 256) {
    const int w   = idx / (K * HID);          // 0 = Wl, 1 = Wr
    const int rem = idx - w * (K * HID);
    const int k = rem / HID, n = rem % HID;
    const int kb = k >> 5, kr = k & 31;
    const int hi = (kr >> 3) & 1, j = (kr & 7) | ((kr >> 4) << 3);
    const int f = w * 4 + (n >> 4);
    const int lane = hi * 16 + (n & 15);
    const float val = w ? Wr[rem] : Wl[rem];
    sW[(((kb * 8 + f) * 32) + lane) * 16 + j] = (_Float16)val;
  }
  __syncthreads();

  const int lane = threadIdx.x & 31;
  const int wave = threadIdx.x >> 5;
  const int tile = blockIdx.x * 8 + wave;
  if (tile >= NND / 16) return;
  const int m0 = tile * 16;
  const int hi = lane >> 4;          // 0: lanes 0-15, 1: lanes 16-31
  const int lrow = m0 + (lane & 15); // A row this lane supplies
  const float* xrow = X + (size_t)lrow * K;

  v8f acc[8] = {v8f{}, v8f{}, v8f{}, v8f{}, v8f{}, v8f{}, v8f{}, v8f{}};

#pragma unroll
  for (int k0 = 0; k0 < K; k0 += 32) {
    // ---- A fragment (16-bit A 16x32 layout):
    // lane<16:  halves 0-7 = K k0..k0+7,   halves 8-15 = K k0+16..k0+23
    // lane>=16: halves 0-7 = K k0+8..+15,  halves 8-15 = K k0+24..+31
    const int kbase = k0 + hi * 8;
    const float4* p = reinterpret_cast<const float4*>(xrow + kbase);
    float4 f0 = p[0], f1 = p[1];
    const float4* q = reinterpret_cast<const float4*>(xrow + kbase + 16);
    float4 f2 = q[0], f3 = q[1];
    v16h a;
    a[0] = (_Float16)f0.x;  a[1] = (_Float16)f0.y;
    a[2] = (_Float16)f0.z;  a[3] = (_Float16)f0.w;
    a[4] = (_Float16)f1.x;  a[5] = (_Float16)f1.y;
    a[6] = (_Float16)f1.z;  a[7] = (_Float16)f1.w;
    a[8] = (_Float16)f2.x;  a[9] = (_Float16)f2.y;
    a[10] = (_Float16)f2.z; a[11] = (_Float16)f2.w;
    a[12] = (_Float16)f3.x; a[13] = (_Float16)f3.y;
    a[14] = (_Float16)f3.z; a[15] = (_Float16)f3.w;

    const int kb = k0 >> 5;
#pragma unroll
    for (int f = 0; f < 8; ++f) {
      // contiguous 32B per-lane B fragment (2x ds_load_b128)
      const v16h b = *reinterpret_cast<const v16h*>(
          &sW[(((kb * 8 + f) * 32) + lane) * 16]);
      acc[f] = __builtin_amdgcn_wmma_f32_16x16x32_f16(
          false, a, false, b, (short)0, acc[f], false, false);
    }
  }

  // ---- store (f32 C/D 16x16 layout): VGPR r, lanes 0-15 -> M=m0+r, N=lane;
  // lanes 16-31 -> M=m0+8+r, N=lane-16
  const int col0 = lane & 15;
  const int rbase = m0 + hi * 8;
#pragma unroll
  for (int nt = 0; nt < 4; ++nt) {
#pragma unroll
    for (int r = 0; r < 8; ++r) {
      Yl[(size_t)(rbase + r) * HID + nt * 16 + col0] = acc[nt][r];
      Yr[(size_t)(rbase + r) * HID + nt * 16 + col0] = acc[4 + nt][r];
    }
  }
}

// ---------------------------------------------------------------- edge scatter
// agg[tgt] += V[src], 64 channels/edge, 1 wave per edge, 2 atomics per lane.
__global__ __launch_bounds__(256) void scatter_kernel(const int* __restrict__ src,
                                                      const int* __restrict__ tgt,
                                                      const float* __restrict__ V,
                                                      float* __restrict__ agg) {
  const int lane = threadIdx.x & 31;
  const int e = blockIdx.x * 8 + (threadIdx.x >> 5);
  if (e >= NEDG) return;
  const int s = src[e];
  const int t = tgt[e];
  const float v0 = V[(size_t)s * HID + lane];
  const float v1 = V[(size_t)s * HID + lane + 32];
  atomicAdd(&agg[(size_t)t * HID + lane], v0);
  atomicAdd(&agg[(size_t)t * HID + lane + 32], v1);
}

// ---------------------------------------------------------------- combine/BN/ReLU
__global__ void combine_kernel(const float* __restrict__ agg,
                               const float* __restrict__ xr,
                               const float* __restrict__ deginv,
                               const float* __restrict__ bl,
                               const float* __restrict__ bng,
                               const float* __restrict__ bnb,
                               const float* __restrict__ bnm,
                               const float* __restrict__ bnv,
                               float* __restrict__ H) {
  const int i = blockIdx.x * blockDim.x + threadIdx.x;
  if (i >= NND * HID) return;
  const int node = i >> 6;
  const int c = i & 63;
  float h = deginv[node] * agg[i] + bl[c] + xr[i];
  h = (h - bnm[c]) * (bng[c] * rsqrtf(bnv[c] + BNEPS)) + bnb[c];
  H[i] = fmaxf(h, 0.0f);
}

// ---------------------------------------------------------------- MLP head
__global__ __launch_bounds__(256) void head_kernel(const float* __restrict__ H,
                                                   const float* __restrict__ Wh1,
                                                   const float* __restrict__ bh1,
                                                   const float* __restrict__ Wh2,
                                                   const float* __restrict__ bh2,
                                                   float* __restrict__ out) {
  __shared__ float sW1[HID * 32];
  __shared__ float sb1[32];
  __shared__ float sW2[32];
  for (int i = threadIdx.x; i < HID * 32; i += 256) sW1[i] = Wh1[i];
  if (threadIdx.x < 32) {
    sb1[threadIdx.x] = bh1[threadIdx.x];
    sW2[threadIdx.x] = Wh2[threadIdx.x];
  }
  __syncthreads();

  const int node = blockIdx.x * 256 + threadIdx.x;
  if (node >= NND) return;

  float hreg[HID];
  const float4* hp = reinterpret_cast<const float4*>(H + (size_t)node * HID);
#pragma unroll
  for (int i = 0; i < 16; ++i) {
    float4 f = hp[i];
    hreg[4 * i + 0] = f.x; hreg[4 * i + 1] = f.y;
    hreg[4 * i + 2] = f.z; hreg[4 * i + 3] = f.w;
  }
  float z = bh2[0];
  for (int j = 0; j < 32; ++j) {
    float s = sb1[j];
#pragma unroll
    for (int k = 0; k < HID; ++k) s = fmaf(hreg[k], sW1[k * 32 + j], s);
    z = fmaf(fmaxf(s, 0.0f), sW2[j], z);
  }
  out[node] = 1.0f / (1.0f + __expf(-z));
}

// ---------------------------------------------------------------- launch
extern "C" void kernel_launch(void* const* d_in, const int* in_sizes, int n_in,
                              void* d_out, int out_size, void* d_ws, size_t ws_size,
                              hipStream_t stream) {
  (void)in_sizes; (void)n_in; (void)out_size; (void)ws_size;
  const float* x   = (const float*)d_in[0];
  const int*   ei  = (const int*)d_in[1];
  const int*   src = ei;            // edge_index[0]
  const int*   tgt = ei + NEDG;     // edge_index[1]
  const float* Wl0 = (const float*)d_in[2];
  const float* bl0 = (const float*)d_in[3];
  const float* Wr0 = (const float*)d_in[4];
  const float* g0  = (const float*)d_in[5];
  const float* b0  = (const float*)d_in[6];
  const float* bm0 = (const float*)d_in[7];
  const float* bv0 = (const float*)d_in[8];
  const float* Wl1 = (const float*)d_in[9];
  const float* bl1 = (const float*)d_in[10];
  const float* Wr1 = (const float*)d_in[11];
  const float* g1  = (const float*)d_in[12];
  const float* b1  = (const float*)d_in[13];
  const float* bm1 = (const float*)d_in[14];
  const float* bv1 = (const float*)d_in[15];
  const float* Wh1 = (const float*)d_in[16];
  const float* bh1 = (const float*)d_in[17];
  const float* Wh2 = (const float*)d_in[18];
  const float* bh2 = (const float*)d_in[19];
  float* out = (float*)d_out;

  // workspace layout (floats)
  float* ws   = (float*)d_ws;
  float* deg  = ws;                                   // N        (then deg_inv)
  float* bufA = deg  + NND;                           // N x 64   XL (aggregated path)
  float* bufB = bufA + (size_t)NND * HID;             // N x 64   XR (self path)
  float* bufC = bufB + (size_t)NND * HID;             // N x 64   agg accumulator
  float* bufH = bufC + (size_t)NND * HID;             // N x 64   hidden activations

  const int NH = NND * HID;
  const int gemmBlocks = (NND / 16 + 7) / 8;          // 782
  const int scatBlocks = NEDG / 8;                    // 200000

  // degree & zeroed accumulator
  zero_f32<<<2048, 256, 0, stream>>>(deg, NND);
  zero_f32<<<8192, 256, 0, stream>>>(bufC, NH);
  degree_kernel<<<(NEDG + 255) / 256, 256, 0, stream>>>(tgt, deg);
  deginv_kernel<<<(NND + 255) / 256, 256, 0, stream>>>(deg);

  // ---- layer 0: project first (linearity of aggregation), single read of x
  gemm2_wmma<INCH><<<gemmBlocks, 256, 0, stream>>>(x, Wl0, Wr0, bufA, bufB);
  scatter_kernel<<<scatBlocks, 256, 0, stream>>>(src, tgt, bufA, bufC);
  combine_kernel<<<(NH + 255) / 256, 256, 0, stream>>>(bufC, bufB, deg, bl0,
                                                       g0, b0, bm0, bv0, bufH);

  // ---- layer 1
  zero_f32<<<8192, 256, 0, stream>>>(bufC, NH);
  gemm2_wmma<HID><<<gemmBlocks, 256, 0, stream>>>(bufH, Wl1, Wr1, bufA, bufB);
  scatter_kernel<<<scatBlocks, 256, 0, stream>>>(src, tgt, bufA, bufC);
  combine_kernel<<<(NH + 255) / 256, 256, 0, stream>>>(bufC, bufB, deg, bl1,
                                                       g1, b1, bm1, bv1, bufH);

  // ---- head
  head_kernel<<<(NND + 255) / 256, 256, 0, stream>>>(bufH, Wh1, bh1, Wh2, bh2, out);
}
